// MultiHeadAttention_2327872274710
// MI455X (gfx1250) — compile-verified, matrix operand order checked by reading, not run on hardware
//
#include <hip/hip_runtime.h>

// ---------------------------------------------------------------------------
// Problem constants (match reference)
// ---------------------------------------------------------------------------
constexpr int BATCH = 4;
constexpr int SEQ   = 2048;
constexpr int DM    = 1024;   // d_model
constexpr int NH    = 16;     // heads
constexpr int HD    = 64;     // head dim

// Workspace layout (elements of __bf16)
constexpr size_t XB_N   = (size_t)BATCH * SEQ * DM;        // 8,388,608
constexpr size_t XB_OFF = 0;
constexpr size_t WT_OFF = XB_OFF + XB_N;                   // 3 transposed weights
constexpr size_t WT_N   = 3ull * DM * DM;
constexpr size_t Q_OFF  = WT_OFF + WT_N;
constexpr size_t QKV_N  = (size_t)BATCH * NH * SEQ * HD;   // == XB_N
constexpr size_t K_OFF  = Q_OFF + QKV_N;
constexpr size_t VT_OFF = K_OFF + QKV_N;

// ---------------------------------------------------------------------------
// Vector types for WMMA
// ---------------------------------------------------------------------------
typedef __bf16 v16bf __attribute__((ext_vector_type(16)));
typedef __bf16 v8bf  __attribute__((ext_vector_type(8)));
typedef __bf16 v4bf  __attribute__((ext_vector_type(4)));
typedef float  v8f   __attribute__((ext_vector_type(8)));

__device__ __forceinline__ v8f wmma_bf16(v16bf a, v16bf b, v8f c) {
  // D = A(16x32 bf16) * B(32x16 bf16) + C(16x16 f32)
  return __builtin_amdgcn_wmma_f32_16x16x32_bf16(
      /*neg_a=*/false, a, /*neg_b=*/false, b,
      /*c_mod=*/(short)0, c, /*reuse_a=*/false, /*reuse_b=*/false);
}

// Load an A/B fragment: per-lane two contiguous 16B groups at p and p+16 elems.
// (16-bit operand layout: elements 0..7 -> K = half*8..half*8+7,
//  elements 8..15 -> K = 16+half*8..; caller bakes `half*8` into p.)
__device__ __forceinline__ v16bf load_frag(const __bf16* p) {
  v8bf lo = *reinterpret_cast<const v8bf*>(p);
  v8bf hi = *reinterpret_cast<const v8bf*>(p + 16);
  union { v8bf h[2]; v16bf v; } u;
  u.h[0] = lo; u.h[1] = hi;
  return u.v;
}

// ---------------------------------------------------------------------------
// Kernel 1: x (f32) -> xb (bf16), vectorized
// ---------------------------------------------------------------------------
__global__ __launch_bounds__(256) void k_cvt_x(const float* __restrict__ x,
                                               __bf16* __restrict__ xb, int n4) {
  int i = blockIdx.x * blockDim.x + threadIdx.x;
  if (i < n4) {
    float4 v = reinterpret_cast<const float4*>(x)[i];
    v4bf o;
    o[0] = (__bf16)v.x; o[1] = (__bf16)v.y;
    o[2] = (__bf16)v.z; o[3] = (__bf16)v.w;
    reinterpret_cast<v4bf*>(xb)[i] = o;
  }
}

// ---------------------------------------------------------------------------
// Kernel 2: W[k][n] (f32) -> Wt[n][k] (bf16)
// ---------------------------------------------------------------------------
__global__ __launch_bounds__(256) void k_wt(const float* __restrict__ Wq,
                                            const float* __restrict__ Wk,
                                            const float* __restrict__ Wv,
                                            __bf16* __restrict__ wt) {
  int idx = blockIdx.x * blockDim.x + threadIdx.x;   // 0 .. DM*DM-1
  int g = blockIdx.y;
  const float* W = (g == 0) ? Wq : ((g == 1) ? Wk : Wv);
  int n = idx >> 10;          // out-feature
  int k = idx & (DM - 1);     // in-feature (contiguous -> coalesced store)
  wt[(size_t)g * DM * DM + (size_t)n * DM + k] = (__bf16)W[(size_t)k * DM + n];
}

// ---------------------------------------------------------------------------
// Kernel 3: QKV projection, ping-pong software pipeline.
// One wave computes a 32x64 output tile (2 M-subtiles share 4 B fragments):
// 8 WMMAs per 6 fragment loads. Buffers alternate roles across the 2x-
// unrolled k-loop so no inter-buffer register copies are generated.
// The final prefetch of each matrix harmlessly reads into the next ws region.
// Q,K stored [B,H,S,HD]; V stored transposed [B,H,HD,S].
// ---------------------------------------------------------------------------
__global__ __launch_bounds__(256) void k_proj(const __bf16* __restrict__ xb,
                                              const __bf16* __restrict__ wt,
                                              __bf16* __restrict__ Qo,
                                              __bf16* __restrict__ Ko,
                                              __bf16* __restrict__ Vt) {
  const int lane = threadIdx.x & 31;
  const int lm   = lane & 15;
  const int half = lane >> 4;
  const int wid  = blockIdx.x * (blockDim.x >> 5) + (threadIdx.x >> 5);

  const int g   = wid >> 12;          // which weight matrix (4096 tiles each)
  const int rem = wid & 4095;
  const int nt  = rem >> 8;           // 16 n-tiles (outer -> shared in block)
  const int mt  = rem & 255;          // 256 m-tiles of 32 rows (inner)
  const int m0  = mt << 5;
  const int n0  = nt << 6;

  const __bf16* Wg = wt + (size_t)g * DM * DM;
  const __bf16* arow[2];
  arow[0] = xb + (size_t)(m0 + lm) * DM + half * 8;
  arow[1] = xb + (size_t)(m0 + 16 + lm) * DM + half * 8;
  const __bf16* brow[4];
#pragma unroll
  for (int j = 0; j < 4; ++j)
    brow[j] = Wg + (size_t)(n0 + j * 16 + lm) * DM + half * 8;

  const v8f zero = {0.f, 0.f, 0.f, 0.f, 0.f, 0.f, 0.f, 0.f};
  v8f acc[2][4];
#pragma unroll
  for (int s = 0; s < 2; ++s)
#pragma unroll
    for (int j = 0; j < 4; ++j) acc[s][j] = zero;

  // prologue: k-step 0 into buffer A
  v16bf aA[2], bA[4], aB[2], bB[4];
#pragma unroll
  for (int s = 0; s < 2; ++s) aA[s] = load_frag(arow[s]);
#pragma unroll
  for (int j = 0; j < 4; ++j) bA[j] = load_frag(brow[j]);

  for (int kk = 0; kk < DM; kk += 64) {
    // prefetch k-step kk+32 into buffer B
#pragma unroll
    for (int s = 0; s < 2; ++s) aB[s] = load_frag(arow[s] + kk + 32);
#pragma unroll
    for (int j = 0; j < 4; ++j) bB[j] = load_frag(brow[j] + kk + 32);
    // compute on buffer A (k-step kk)
#pragma unroll
    for (int j = 0; j < 4; ++j) {
      acc[0][j] = wmma_bf16(aA[0], bA[j], acc[0][j]);
      acc[1][j] = wmma_bf16(aA[1], bA[j], acc[1][j]);
    }
    // prefetch k-step kk+64 into buffer A (dead on last iter, stays in ws)
#pragma unroll
    for (int s = 0; s < 2; ++s) aA[s] = load_frag(arow[s] + kk + 64);
#pragma unroll
    for (int j = 0; j < 4; ++j) bA[j] = load_frag(brow[j] + kk + 64);
    // compute on buffer B (k-step kk+32)
#pragma unroll
    for (int j = 0; j < 4; ++j) {
      acc[0][j] = wmma_bf16(aB[0], bB[j], acc[0][j]);
      acc[1][j] = wmma_bf16(aB[1], bB[j], acc[1][j]);
    }
  }

  // C/D layout: lanes 0-15 hold M=r, lanes 16-31 hold M=8+r; N = lane%16
#pragma unroll
  for (int s = 0; s < 2; ++s) {
    const int rbase = m0 + s * 16 + 8 * half;
#pragma unroll
    for (int j = 0; j < 4; ++j) {
      const int col = n0 + j * 16 + lm;
      const int h   = col >> 6;
      const int hd  = col & (HD - 1);
#pragma unroll
      for (int r = 0; r < 8; ++r) {
        const int row = rbase + r;
        const int b   = row >> 11;         // / SEQ
        const int sq  = row & (SEQ - 1);
        const size_t bh = (size_t)(b * NH + h);
        const __bf16 v = (__bf16)acc[s][j][r];
        if (g == 0)      Qo[(bh * SEQ + sq) * HD + hd] = v;
        else if (g == 1) Ko[(bh * SEQ + sq) * HD + hd] = v;
        else             Vt[(bh * HD + hd) * SEQ + sq] = v;
      }
    }
  }
}

// ---------------------------------------------------------------------------
// Kernel 4: causal flash attention. One wave per (b,h, 32-query tile).
// Per 32-key chunk: 8 score WMMAs, online softmax, P relayout via LDS,
// 8 PV WMMAs. V fragments load early (hide under softmax); next chunk's
// K/V cachelines are prefetched at the top of each iteration.
// ---------------------------------------------------------------------------
__global__ __launch_bounds__(32) void k_attn(const __bf16* __restrict__ Q,
                                             const __bf16* __restrict__ K,
                                             const __bf16* __restrict__ Vt,
                                             float* __restrict__ out) {
  __shared__ __bf16 Pb[32][32];   // P tiles (2 M-tiles x 32 keys), D->A relayout

  const int lane = threadIdx.x & 31;
  const int lm   = lane & 15;
  const int half = lane >> 4;

  const int q0 = blockIdx.x * 32;       // 32 queries per wave
  const int bh = blockIdx.y;            // b*NH + h
  const int b  = bh >> 4;
  const int h  = bh & (NH - 1);

  const __bf16* Qb = Q  + (size_t)bh * SEQ * HD;
  const __bf16* Kb = K  + (size_t)bh * SEQ * HD;
  const __bf16* Vb = Vt + (size_t)bh * HD * SEQ;

  // Q fragments: 2 M-tiles x 2 k-steps (HD = 64)
  v16bf qa[2][2];
#pragma unroll
  for (int mt = 0; mt < 2; ++mt)
#pragma unroll
    for (int ks = 0; ks < 2; ++ks)
      qa[mt][ks] =
          load_frag(Qb + (size_t)(q0 + mt * 16 + lm) * HD + ks * 32 + half * 8);

  const v8f zero = {0.f, 0.f, 0.f, 0.f, 0.f, 0.f, 0.f, 0.f};
  v8f accO[2][4];
#pragma unroll
  for (int mt = 0; mt < 2; ++mt)
#pragma unroll
    for (int j = 0; j < 4; ++j) accO[mt][j] = zero;

  float mrun[2][8], lrun[2][8];
#pragma unroll
  for (int mt = 0; mt < 2; ++mt)
#pragma unroll
    for (int r = 0; r < 8; ++r) { mrun[mt][r] = -3.0e38f; lrun[mt][r] = 0.f; }

  // softmax over scores/sqrt(HD): fold 1/8 * log2(e) into exp2
  const float ls = 0.18033688011112042f;

  for (int c0 = 0; c0 < q0 + 32; c0 += 32) {
    // ---- prefetch next chunk's K rows and V rows (one line per lane) ----
    if (c0 < q0) {
      __builtin_prefetch(Kb + (size_t)(c0 + 32 + lane) * HD, 0, 1);
      __builtin_prefetch(Vb + (size_t)lane * SEQ + c0 + 32, 0, 1);
      __builtin_prefetch(Vb + (size_t)(lane + 32) * SEQ + c0 + 32, 0, 1);
    }

    // ---- K fragments for this 32-key chunk (2 N-tiles x 2 k-steps) ----
    v16bf kf[2][2];
#pragma unroll
    for (int nt = 0; nt < 2; ++nt)
#pragma unroll
      for (int ks = 0; ks < 2; ++ks)
        kf[nt][ks] = load_frag(Kb + (size_t)(c0 + nt * 16 + lm) * HD +
                               ks * 32 + half * 8);
    // ---- V fragments issued early: latency hides under softmax ----
    v16bf vf[4];
#pragma unroll
    for (int j = 0; j < 4; ++j)
      vf[j] = load_frag(Vb + (size_t)(j * 16 + lm) * SEQ + c0 + half * 8);

    // ---- scores: 32 queries x 32 keys ----
    v8f d[2][2];
#pragma unroll
    for (int mt = 0; mt < 2; ++mt)
#pragma unroll
      for (int nt = 0; nt < 2; ++nt) {
        v8f t = wmma_bf16(qa[mt][0], kf[nt][0], zero);
        d[mt][nt] = wmma_bf16(qa[mt][1], kf[nt][1], t);
      }

    // ---- causal mask (only the trailing chunk can have masked entries) ----
    if (c0 + 31 > q0) {
#pragma unroll
      for (int mt = 0; mt < 2; ++mt)
#pragma unroll
        for (int r = 0; r < 8; ++r) {
          const int q = q0 + mt * 16 + r + 8 * half;   // this lane's row
          if (c0 + lm > q)      d[mt][0][r] = -3.0e38f;
          if (c0 + 16 + lm > q) d[mt][1][r] = -3.0e38f;
        }
    }

    // ---- online softmax; write P straight into LDS ----
    __syncthreads();   // protect previous chunk's P reads (nop for 1 wave)
#pragma unroll
    for (int mt = 0; mt < 2; ++mt)
#pragma unroll
      for (int r = 0; r < 8; ++r) {
        float t = fmaxf(d[mt][0][r], d[mt][1][r]);
        t = fmaxf(t, __shfl_xor(t, 1, 32));
        t = fmaxf(t, __shfl_xor(t, 2, 32));
        t = fmaxf(t, __shfl_xor(t, 4, 32));
        t = fmaxf(t, __shfl_xor(t, 8, 32));
        const float mnew  = fmaxf(mrun[mt][r], t);
        const float alpha = exp2f((mrun[mt][r] - mnew) * ls);
        mrun[mt][r] = mnew;
        const float e0 = exp2f((d[mt][0][r] - mnew) * ls);
        const float e1 = exp2f((d[mt][1][r] - mnew) * ls);
        const int M = mt * 16 + r + 8 * half;
        Pb[M][lm]      = (__bf16)e0;
        Pb[M][16 + lm] = (__bf16)e1;
        float rs = e0 + e1;
        rs += __shfl_xor(rs, 1, 32);
        rs += __shfl_xor(rs, 2, 32);
        rs += __shfl_xor(rs, 4, 32);
        rs += __shfl_xor(rs, 8, 32);
        lrun[mt][r] = lrun[mt][r] * alpha + rs;
#pragma unroll
        for (int j = 0; j < 4; ++j) accO[mt][j][r] *= alpha;
      }
    __syncthreads();

    // ---- P back as A-fragments, then PV ----
    v16bf pa[2];
#pragma unroll
    for (int mt = 0; mt < 2; ++mt)
      pa[mt] = load_frag(&Pb[mt * 16 + lm][half * 8]);
#pragma unroll
    for (int mt = 0; mt < 2; ++mt)
#pragma unroll
      for (int j = 0; j < 4; ++j)
        accO[mt][j] = wmma_bf16(pa[mt], vf[j], accO[mt][j]);
  }

  // ---- epilogue: divide by row sums, store fp32 [B,S,DM] ----
#pragma unroll
  for (int mt = 0; mt < 2; ++mt)
#pragma unroll
    for (int r = 0; r < 8; ++r) {
      const float inv = 1.0f / lrun[mt][r];
      const int   row = q0 + mt * 16 + r + 8 * half;
      float* orow = out + ((size_t)b * SEQ + row) * DM + h * HD;
#pragma unroll
      for (int j = 0; j < 4; ++j)
        orow[j * 16 + lm] = accO[mt][j][r] * inv;
    }
}

// ---------------------------------------------------------------------------
// Host-side launcher
// ---------------------------------------------------------------------------
extern "C" void kernel_launch(void* const* d_in, const int* in_sizes, int n_in,
                              void* d_out, int out_size, void* d_ws, size_t ws_size,
                              hipStream_t stream) {
  const float* x  = (const float*)d_in[0];
  const float* Wq = (const float*)d_in[1];
  const float* Wk = (const float*)d_in[2];
  const float* Wv = (const float*)d_in[3];
  float* out = (float*)d_out;

  __bf16* wsb = (__bf16*)d_ws;
  __bf16* xb  = wsb + XB_OFF;
  __bf16* wt  = wsb + WT_OFF;
  __bf16* Qo  = wsb + Q_OFF;
  __bf16* Ko  = wsb + K_OFF;
  __bf16* Vt  = wsb + VT_OFF;

  // 1) x -> bf16
  {
    int n4 = (int)(XB_N / 4);
    k_cvt_x<<<(n4 + 255) / 256, 256, 0, stream>>>(x, xb, n4);
  }
  // 2) W -> bf16 transposed
  {
    dim3 grid((DM * DM) / 256, 3);
    k_wt<<<grid, 256, 0, stream>>>(Wq, Wk, Wv, wt);
  }
  // 3) QKV projection: 3 * 16 * 256 = 12288 waves, 8 waves/block
  {
    k_proj<<<12288 / 8, 256, 0, stream>>>(xb, wt, Qo, Ko, Vt);
  }
  // 4) flash attention: (S/32) x (B*H) waves
  {
    dim3 grid(SEQ / 32, BATCH * NH);
    k_attn<<<grid, 32, 0, stream>>>(Qo, Ko, Vt, out);
  }
}
